// UncertaintyGAT_24077586662039
// MI455X (gfx1250) — matrix operand order repaired and checked by reading, not compiled, not run on hardware
//
#include <hip/hip_runtime.h>
#include <hip/hip_bf16.h>

#define N_NODES 100000
#define N_EDGES 1600000
#define F_DIM   64

typedef __attribute__((ext_vector_type(16))) __bf16 bf16x16;
typedef __attribute__((ext_vector_type(8)))  __bf16 bf16x8;
typedef __attribute__((ext_vector_type(8)))  float  f32x8;

// ---------- helpers ----------
static __device__ __forceinline__ unsigned short f32_to_bf16_rne(float f) {
  unsigned u = __float_as_uint(f);
  unsigned r = u + 0x7fffu + ((u >> 16) & 1u);
  return (unsigned short)(r >> 16);
}
// order-preserving float->uint encoding for atomicMax-based segment max
static __device__ __forceinline__ unsigned enc_ord(float f) {
  unsigned u = __float_as_uint(f);
  return (u & 0x80000000u) ? ~u : (u | 0x80000000u);
}
static __device__ __forceinline__ float dec_ord(unsigned u) {
  return (u & 0x80000000u) ? __uint_as_float(u ^ 0x80000000u) : __uint_as_float(~u);
}
static __device__ __forceinline__ float lrelu(float x) { return x > 0.f ? x : 0.2f * x; }

// ---------- conversion kernels ----------
__global__ void k_f32_to_bf16(const float* __restrict__ in,
                              unsigned short* __restrict__ out, int n) {
  int i = blockIdx.x * blockDim.x + threadIdx.x;
  if (i < n) out[i] = f32_to_bf16_rne(in[i]);
}

__global__ void k_relu_to_bf16(const float* __restrict__ in,
                               unsigned short* __restrict__ out, int n) {
  int i = blockIdx.x * blockDim.x + threadIdx.x;
  if (i < n) out[i] = f32_to_bf16_rne(fmaxf(in[i], 0.f));
}

// W [K,Nout] row-major -> Wt [Nout,K] bf16 (so WMMA B-operand loads are contiguous)
__global__ void k_w_transpose_bf16(const float* __restrict__ W,
                                   unsigned short* __restrict__ Wt, int K, int Nout) {
  int i = blockIdx.x * blockDim.x + threadIdx.x;
  if (i < K * Nout) {
    int k = i / Nout, n = i % Nout;
    Wt[n * K + k] = f32_to_bf16_rne(W[i]);
  }
}

// ---------- WMMA GEMM: Z[N,64] = Xb[N,64] @ W[64,64]  (Wt = W^T, bf16) ----------
__global__ __launch_bounds__(128) void gat_gemm_bf16_wmma(
    const __bf16* __restrict__ Xb,   // [N,64] bf16 row-major
    const __bf16* __restrict__ Wt,   // [64(nout),64(k)] bf16
    float* __restrict__ Z) {         // [N,64] fp32
  const int lane = threadIdx.x & 31;
  const int wave = threadIdx.x >> 5;           // 4 waves -> 4 col tiles of 16
  const int rowb = blockIdx.x * 16;
  const int row  = rowb + (lane & 15);
  const int colb = wave * 16;
  const int koff = (lane >> 4) * 16;           // A-operand half-wave K offset
  const int ncol = colb + (lane & 15);         // B-operand column per lane

  __builtin_prefetch(Xb + row * F_DIM + 32, 0, 1);  // global_prefetch_b8

  f32x8 acc = {};
#pragma unroll
  for (int ks = 0; ks < 2; ++ks) {             // K=64 = 2 x 32
    const int kb = ks * 32;
    // A 16x32 bf16: VGPRs0-3 hold K=koff..koff+7, VGPRs4-7 hold K=koff+8..koff+15
    bf16x8 a_lo = *(const bf16x8*)(Xb + row * F_DIM + kb + koff);
    bf16x8 a_hi = *(const bf16x8*)(Xb + row * F_DIM + kb + 8 + koff);
    bf16x16 a;
#pragma unroll
    for (int i = 0; i < 8; ++i) { a[i] = a_lo[i]; a[i + 8] = a_hi[i]; }
    // B 32x16 bf16: lane holds 16 consecutive K at its column
    const int kb2 = kb + (lane >> 4) * 16;
    bf16x16 b = *(const bf16x16*)(Wt + ncol * F_DIM + kb2);
    acc = __builtin_amdgcn_wmma_f32_16x16x32_bf16(false, a, false, b,
                                                  (short)0, acc, false, false);
  }
  // C/D layout: VGPR v -> row m0+v, lane&15 -> col
  const int m0 = (lane < 16) ? 0 : 8;
#pragma unroll
  for (int v = 0; v < 8; ++v)
    Z[(rowb + m0 + v) * F_DIM + colb + (lane & 15)] = acc[v];
}

// ---------- attention kernels (generic over H, D) ----------
__global__ void k_el_er(const float* __restrict__ Z, const float* __restrict__ al,
                        const float* __restrict__ ar, float* __restrict__ el,
                        float* __restrict__ er, int n, int H, int D) {
  int i = blockIdx.x * blockDim.x + threadIdx.x;
  if (i >= n * H) return;
  int h = i % H;
  const float* z = Z + (size_t)i * D;
  float sl = 0.f, sr = 0.f;
  for (int d = 0; d < D; ++d) { float zv = z[d]; sl += zv * al[h * D + d]; sr += zv * ar[h * D + d]; }
  el[i] = sl; er[i] = sr;
}

__global__ void k_init(float* __restrict__ O, const float* __restrict__ b,
                       unsigned* __restrict__ mx, float* __restrict__ den,
                       int n, int H, int D) {
  int i = blockIdx.x * blockDim.x + threadIdx.x;
  if (i < n * H * D) O[i] = b[i % (H * D)];
  if (i < n * H) { mx[i] = 0u; den[i] = 0.f; }
}

__global__ void k_edge_max(const int* __restrict__ src, const int* __restrict__ dst,
                           const float* __restrict__ el, const float* __restrict__ er,
                           unsigned* __restrict__ mx, int E, int H) {
  int e = blockIdx.x * blockDim.x + threadIdx.x;
  if (e >= E) return;
  int s = src[e], d = dst[e];
  for (int h = 0; h < H; ++h) {
    float sc = lrelu(el[s * H + h] + er[d * H + h]);
    atomicMax(&mx[d * H + h], enc_ord(sc));
  }
}

__global__ void k_edge_den(const int* __restrict__ src, const int* __restrict__ dst,
                           const float* __restrict__ el, const float* __restrict__ er,
                           const unsigned* __restrict__ mx, float* __restrict__ den,
                           int E, int H) {
  int e = blockIdx.x * blockDim.x + threadIdx.x;
  if (e >= E) return;
  int s = src[e], d = dst[e];
  for (int h = 0; h < H; ++h) {
    float sc = lrelu(el[s * H + h] + er[d * H + h]);
    float m  = dec_ord(mx[d * H + h]);
    atomicAdd(&den[d * H + h], __expf(sc - m));
  }
}

__global__ void k_edge_agg(const int* __restrict__ src, const int* __restrict__ dst,
                           const float* __restrict__ el, const float* __restrict__ er,
                           const unsigned* __restrict__ mx, const float* __restrict__ den,
                           const float* __restrict__ Z, float* __restrict__ O,
                           int E, int H, int D) {
  int t = blockIdx.x * blockDim.x + threadIdx.x;
  if (t >= E * H) return;
  int e = t / H, h = t % H;
  int s = src[e], d = dst[e];
  float sc = lrelu(el[s * H + h] + er[d * H + h]);
  float m  = dec_ord(mx[d * H + h]);
  float alpha = __expf(sc - m) / fmaxf(den[d * H + h], 1e-9f);
  const float* zp = Z + (size_t)(s * H + h) * D;
  float* op = O + (size_t)(d * H + h) * D;
  for (int dd = 0; dd < D; ++dd) atomicAdd(&op[dd], alpha * zp[dd]);
}

// final layer projection: Zo[n] = sum_k relu(Hb[n,k]) * Wo[k]
__global__ void k_final_dot(const float* __restrict__ Hb, const float* __restrict__ Wo,
                            float* __restrict__ Zo, int n) {
  int i = blockIdx.x * blockDim.x + threadIdx.x;
  if (i >= n) return;
  float s = 0.f;
  for (int k = 0; k < F_DIM; ++k) s += fmaxf(Hb[i * F_DIM + k], 0.f) * Wo[k];
  Zo[i] = s;
}

// ---------- launch ----------
extern "C" void kernel_launch(void* const* d_in, const int* in_sizes, int n_in,
                              void* d_out, int out_size, void* d_ws, size_t ws_size,
                              hipStream_t stream) {
  (void)in_sizes; (void)n_in; (void)out_size; (void)ws_size;
  const float* feat = (const float*)d_in[0];
  const int*   src  = (const int*)d_in[1];
  const int*   dst  = (const int*)d_in[2];
  const float* Ws[2]  = {(const float*)d_in[3], (const float*)d_in[7]};
  const float* als[2] = {(const float*)d_in[4], (const float*)d_in[8]};
  const float* ars[2] = {(const float*)d_in[5], (const float*)d_in[9]};
  const float* bs[2]  = {(const float*)d_in[6], (const float*)d_in[10]};
  const float* Wo  = (const float*)d_in[11];
  const float* alo = (const float*)d_in[12];
  const float* aro = (const float*)d_in[13];
  const float* bo  = (const float*)d_in[14];
  float* out = (float*)d_out;

  char* ws = (char*)d_ws;
  size_t off = 0;
  auto alloc = [&](size_t bytes) -> void* {
    void* p = ws + off;
    off += (bytes + 255) & ~(size_t)255;
    return p;
  };
  float*          Z   = (float*)alloc((size_t)N_NODES * F_DIM * 4);
  float*          Hb  = (float*)alloc((size_t)N_NODES * F_DIM * 4);
  unsigned short* Xb  = (unsigned short*)alloc((size_t)N_NODES * F_DIM * 2);
  unsigned short* Wt  = (unsigned short*)alloc((size_t)F_DIM * F_DIM * 2);
  float*          EL  = (float*)alloc((size_t)N_NODES * 4 * 4);
  float*          ER  = (float*)alloc((size_t)N_NODES * 4 * 4);
  unsigned*       MX  = (unsigned*)alloc((size_t)N_NODES * 4 * 4);
  float*          DEN = (float*)alloc((size_t)N_NODES * 4 * 4);

  const int T = 256;
  const int NF  = N_NODES * F_DIM;
  const int NH4 = N_NODES * 4;

  k_f32_to_bf16<<<(NF + T - 1) / T, T, 0, stream>>>(feat, Xb, NF);

  for (int L = 0; L < 2; ++L) {   // two hidden GAT layers: H=4, D=16
    k_w_transpose_bf16<<<(F_DIM * F_DIM + T - 1) / T, T, 0, stream>>>(Ws[L], Wt, F_DIM, F_DIM);
    gat_gemm_bf16_wmma<<<N_NODES / 16, 128, 0, stream>>>(
        (const __bf16*)Xb, (const __bf16*)Wt, Z);
    k_el_er<<<(NH4 + T - 1) / T, T, 0, stream>>>(Z, als[L], ars[L], EL, ER, N_NODES, 4, 16);
    k_init<<<(NF + T - 1) / T, T, 0, stream>>>(Hb, bs[L], MX, DEN, N_NODES, 4, 16);
    k_edge_max<<<(N_EDGES + T - 1) / T, T, 0, stream>>>(src, dst, EL, ER, MX, N_EDGES, 4);
    k_edge_den<<<(N_EDGES + T - 1) / T, T, 0, stream>>>(src, dst, EL, ER, MX, DEN, N_EDGES, 4);
    k_edge_agg<<<(N_EDGES * 4 + T - 1) / T, T, 0, stream>>>(src, dst, EL, ER, MX, DEN, Z, Hb,
                                                           N_EDGES, 4, 16);
    if (L == 0)
      k_relu_to_bf16<<<(NF + T - 1) / T, T, 0, stream>>>(Hb, Xb, NF);
  }

  // output layer: H=1, D=1 (z reuses Z buffer)
  k_final_dot<<<(N_NODES + T - 1) / T, T, 0, stream>>>(Hb, Wo, Z, N_NODES);
  k_el_er<<<(N_NODES + T - 1) / T, T, 0, stream>>>(Z, alo, aro, EL, ER, N_NODES, 1, 1);
  k_init<<<(N_NODES + T - 1) / T, T, 0, stream>>>(out, bo, MX, DEN, N_NODES, 1, 1);
  k_edge_max<<<(N_EDGES + T - 1) / T, T, 0, stream>>>(src, dst, EL, ER, MX, N_EDGES, 1);
  k_edge_den<<<(N_EDGES + T - 1) / T, T, 0, stream>>>(src, dst, EL, ER, MX, DEN, N_EDGES, 1);
  k_edge_agg<<<(N_EDGES + T - 1) / T, T, 0, stream>>>(src, dst, EL, ER, MX, DEN, Z, out,
                                                     N_EDGES, 1, 1);
}
